// Raindrop_v4_58531814310313
// MI455X (gfx1250) — matrix operand order repaired and checked by reading, not compiled, not run on hardware
//
#include <hip/hip_runtime.h>
#include <hip/hip_bf16.h>
#include <math.h>

// ---------------- model dims ----------------
#define T_LEN   215
#define B_SZ    512
#define D_INP   36
#define D_OB    4
#define D_PE    16
#define NHEAD   4
#define NHID    256
#define DM      720          // D_INP * (D_OB + 16)
#define HD      180          // DM / NHEAD
#define M_TOK   (T_LEN*B_SZ) // 110080
#define CC      860          // T * D_OB (graph feature dim)
#define E0      1296
#define E1      648
#define TPAD    224          // T padded to mult of 16 (and 32 for K-dim use)
#define HDPAD   192          // HD padded to mult of 32
#define KP_DM   736          // 720 padded to mult of 32
#define DFIN    756          // D_INP*(D_OB+D_PE+1)
#define DFPAD   768

typedef __bf16 bf16_t;
typedef bf16_t v16bf __attribute__((ext_vector_type(16)));
typedef bf16_t v8bf  __attribute__((ext_vector_type(8)));
typedef float  v8f   __attribute__((ext_vector_type(8)));

// ---------------- small helpers ----------------
__device__ __forceinline__ float block_reduce_sum256(float v, float* red) {
    int tid = threadIdx.x;
    red[tid] = v; __syncthreads();
    for (int s = 128; s > 0; s >>= 1) { if (tid < s) red[tid] += red[tid + s]; __syncthreads(); }
    float r = red[0]; __syncthreads();
    return r;
}

// ======================================================================
// Elementwise / reshaping kernels
// ======================================================================

// h = relu(repeat(xsrc,4,-1) * R_u)   h:(T,B,144)
__global__ void h_kernel(const float* __restrict__ src, const float* __restrict__ R_u,
                         float* __restrict__ h, size_t total) {
    size_t i = (size_t)blockIdx.x * 256 + threadIdx.x;
    if (i >= total) return;
    int j = (int)(i % 144);
    size_t tb = i / 144;
    float v = src[tb * 72 + (j >> 2)] * R_u[j];
    h[i] = fmaxf(v, 0.f);
}

// G0[b,n, t*4+o] = h[t,b, n*4+o]
__global__ void g0_kernel(const float* __restrict__ h, float* __restrict__ G0, size_t total) {
    size_t i = (size_t)blockIdx.x * 256 + threadIdx.x;
    if (i >= total) return;
    int f = (int)(i % CC);
    int n = (int)((i / CC) % D_INP);
    int b = (int)(i / ((size_t)CC * D_INP));
    int t = f >> 2, o = f & 3;
    G0[i] = h[((size_t)t * B_SZ + b) * 144 + n * 4 + o];
}

// graphout[t,b, n*4+o] = G2[b,n, t*4+o]
__global__ void gout_kernel(const float* __restrict__ G2, float* __restrict__ gout, size_t total) {
    size_t i = (size_t)blockIdx.x * 256 + threadIdx.x;
    if (i >= total) return;
    int j = (int)(i % 144);
    size_t tb = i / 144;
    int t = (int)(tb / B_SZ), b = (int)(tb % B_SZ);
    int n = j >> 2, o = j & 3;
    gout[i] = G2[((size_t)b * D_INP + n) * CC + t * 4 + o];
}

// positional encoding (T,B,16)
__global__ void pe_kernel(const float* __restrict__ times, float* __restrict__ pe, size_t total) {
    size_t i = (size_t)blockIdx.x * 256 + threadIdx.x;
    if (i >= total) return;
    int j = (int)(i % 16);
    size_t tb = i / 16;
    float ts = powf((float)T_LEN, (float)(j & 7) / 7.0f);
    float sv = times[tb] / ts;
    pe[i] = (j < 8) ? sinf(sv) : cosf(sv);
}

// emb = static @ emb_W.T + emb_b   (B,36)
__global__ void emb_kernel(const float* __restrict__ st, const float* __restrict__ W,
                           const float* __restrict__ bvec, float* __restrict__ emb) {
    int i = blockIdx.x * 256 + threadIdx.x;
    if (i >= B_SZ * D_INP) return;
    int b = i / D_INP, n = i % D_INP;
    float s = bvec[n];
    for (int d = 0; d < 9; ++d) s += st[b * 9 + d] * W[n * 9 + d];
    emb[i] = s;
}

// X[t,b, n*20+c] = c<4 ? graphout[t,b,n*4+c] : pe[t,b,c-4]
__global__ void buildx_kernel(const float* __restrict__ gout, const float* __restrict__ pe,
                              float* __restrict__ X, size_t total) {
    size_t i = (size_t)blockIdx.x * 256 + threadIdx.x;
    if (i >= total) return;
    int c = (int)(i % DM);
    size_t tb = i / DM;
    int n = c / 20, cc = c % 20;
    X[i] = (cc < 4) ? gout[tb * 144 + n * 4 + cc] : pe[tb * 16 + (cc - 4)];
}

// generic f32 -> bf16 pack with zero padding in both dims
__global__ void pack_pad2d(const float* __restrict__ src, bf16_t* __restrict__ dst,
                           int Msrc, int Ksrc, int Kdst, size_t total) {
    size_t i = (size_t)blockIdx.x * 256 + threadIdx.x;
    if (i >= total) return;
    int k = (int)(i % Kdst);
    size_t m = i / Kdst;
    float v = (m < (size_t)Msrc && k < Ksrc) ? src[m * Ksrc + k] : 0.f;
    dst[i] = (bf16_t)v;
}

// pack Q/K rows: dst[bh][t][k] = Q[(t*B+b)*720 + h*180 + k], zero-padded
__global__ void pack_qk(const float* __restrict__ Q, bf16_t* __restrict__ dst, size_t total) {
    size_t i = (size_t)blockIdx.x * 256 + threadIdx.x;
    if (i >= total) return;
    int k = (int)(i % HDPAD);
    int t = (int)((i / HDPAD) % TPAD);
    int bh = (int)(i / ((size_t)HDPAD * TPAD));
    int b = bh >> 2, hh = bh & 3;
    float v = (t < T_LEN && k < HD) ? Q[((size_t)t * B_SZ + b) * DM + hh * HD + k] : 0.f;
    dst[i] = (bf16_t)v;
}

// pack V transposed: dst[bh][d][s] = V[(s*B+b)*720 + h*180 + d], zero-padded
__global__ void pack_vt(const float* __restrict__ V, bf16_t* __restrict__ dst, size_t total) {
    size_t i = (size_t)blockIdx.x * 256 + threadIdx.x;
    if (i >= total) return;
    int s = (int)(i % TPAD);
    int d = (int)((i / TPAD) % HDPAD);
    int bh = (int)(i / ((size_t)TPAD * HDPAD));
    int b = bh >> 2, hh = bh & 3;
    float v = (s < T_LEN && d < HD) ? V[((size_t)s * B_SZ + b) * DM + hh * HD + d] : 0.f;
    dst[i] = (bf16_t)v;
}

// ======================================================================
// WMMA GEMM:  C[M,N] = act( A[M,Kp](bf16) @ Bw[Np,Kp](bf16)^T + bias[N] )
// one wave per 16x16 C tile; 8 waves / block cover 128 rows of M.
// ======================================================================
__global__ __launch_bounds__(256)
void gemm_bf16_nt(const bf16_t* __restrict__ A, const bf16_t* __restrict__ Bw,
                  const float* __restrict__ bias, float* __restrict__ Cout,
                  int M, int N, int Kp, int relu) {
    int lane = threadIdx.x & 31;
    int wv   = threadIdx.x >> 5;
    int m0 = (blockIdx.x * 8 + wv) * 16;
    int n0 = blockIdx.y * 16;
    if (m0 >= M) return;                        // wave-uniform
    int rowA  = m0 + (lane & 15);
    int kgrpA = (lane >> 4) * 16;
    int colB  = n0 + (lane & 15);
    int kgrpB = (lane >> 4) * 16;
    const bf16_t* Ab = A  + (size_t)rowA * Kp;
    const bf16_t* Bb = Bw + (size_t)colB * Kp;
    v8f acc = {0.f, 0.f, 0.f, 0.f, 0.f, 0.f, 0.f, 0.f};
    int ksteps = Kp >> 5;
    for (int kt = 0; kt < ksteps; ++kt) {
        int kb = kt * 32 + kgrpA;
        v8bf lo = *(const v8bf*)(Ab + kb);
        v8bf hi = *(const v8bf*)(Ab + kb + 8);
        v16bf afrag = __builtin_shufflevector(lo, hi, 0,1,2,3,4,5,6,7,8,9,10,11,12,13,14,15);
        v16bf bfrag = *(const v16bf*)(Bb + kt * 32 + kgrpB);
        acc = __builtin_amdgcn_wmma_f32_16x16x32_bf16(false, afrag, false, bfrag,
                                                      (short)0, acc, false, false);
    }
    int col = n0 + (lane & 15);
    int rb  = (lane >> 4) * 8;
    if (col < N) {
        float bv = bias ? bias[col] : 0.f;
        for (int r = 0; r < 8; ++r) {
            int row = m0 + rb + r;
            if (row < M) {
                float v = acc[r] + bv;
                if (relu) v = fmaxf(v, 0.f);
                Cout[(size_t)row * N + col] = v;
            }
        }
    }
}

// ======================================================================
// Fused attention: one wave per (b,h, 16-row t-tile).
// scores = QK^T/sqrt(180), mask, softmax, out = P @ V^T
// ======================================================================
__global__ __launch_bounds__(32)
void attn_kernel(const bf16_t* __restrict__ Qp, const bf16_t* __restrict__ Kp,
                 const bf16_t* __restrict__ VTp, const int* __restrict__ lengths,
                 float* __restrict__ Out) {
    int bh = blockIdx.x;           // b*4 + h
    int tt = blockIdx.y;           // 0..13
    int b = bh >> 2, hh = bh & 3;
    int lane = threadIdx.x;
    int len = lengths[b];
    __shared__ float  sc[16 * TPAD];
    __shared__ bf16_t pb[16 * TPAD];
    const bf16_t* Qb = Qp  + (size_t)bh * TPAD * HDPAD;
    const bf16_t* Kb = Kp  + (size_t)bh * TPAD * HDPAD;
    const bf16_t* Vb = VTp + (size_t)bh * HDPAD * TPAD;
    int t0 = tt * 16;

    // preload Q fragments (6 k-steps over HDPAD=192)
    v16bf qf[6];
    {
        int row = t0 + (lane & 15);
        int kg  = (lane >> 4) * 16;
        for (int ks = 0; ks < 6; ++ks) {
            const bf16_t* p = Qb + (size_t)row * HDPAD + ks * 32 + kg;
            v8bf lo = *(const v8bf*)p;
            v8bf hi = *(const v8bf*)(p + 8);
            qf[ks] = __builtin_shufflevector(lo, hi, 0,1,2,3,4,5,6,7,8,9,10,11,12,13,14,15);
        }
    }
    const float scale = 0.074535599249993f;   // 1/sqrt(180)

    // scores into LDS
    int kgB = (lane >> 4) * 16;
    for (int st = 0; st < TPAD / 16; ++st) {
        v8f acc = {0.f, 0.f, 0.f, 0.f, 0.f, 0.f, 0.f, 0.f};
        int colK = st * 16 + (lane & 15);
        for (int ks = 0; ks < 6; ++ks) {
            v16bf kfrag = *(const v16bf*)(Kb + (size_t)colK * HDPAD + ks * 32 + kgB);
            acc = __builtin_amdgcn_wmma_f32_16x16x32_bf16(false, qf[ks], false, kfrag,
                                                          (short)0, acc, false, false);
        }
        int s  = st * 16 + (lane & 15);
        int rb = (lane >> 4) * 8;
        bool masked = (s >= len) || (s >= T_LEN);
        for (int r = 0; r < 8; ++r) {
            float v = acc[r] * scale;
            sc[(rb + r) * TPAD + s] = masked ? -1.0e9f : v;
        }
    }
    __syncthreads();

    // row softmax (wave32 reductions)
    for (int r = 0; r < 16; ++r) {
        float mx = -1.0e30f;
        for (int s = lane; s < TPAD; s += 32) mx = fmaxf(mx, sc[r * TPAD + s]);
        for (int off = 16; off; off >>= 1) mx = fmaxf(mx, __shfl_xor(mx, off, 32));
        float den = 0.f;
        for (int s = lane; s < TPAD; s += 32) den += expf(sc[r * TPAD + s] - mx);
        for (int off = 16; off; off >>= 1) den += __shfl_xor(den, off, 32);
        float inv = 1.f / den;
        for (int s = lane; s < TPAD; s += 32)
            pb[r * TPAD + s] = (bf16_t)(expf(sc[r * TPAD + s] - mx) * inv);
    }
    __syncthreads();

    // out = P(16 x 224) @ V^T  -> (16 x 180)
    int rowP = lane & 15;
    int kgA  = (lane >> 4) * 16;
    for (int dt = 0; dt < HDPAD / 16; ++dt) {
        v8f acc = {0.f, 0.f, 0.f, 0.f, 0.f, 0.f, 0.f, 0.f};
        int colD = dt * 16 + (lane & 15);
        for (int ks = 0; ks < TPAD / 32; ++ks) {
            v16bf af;
            int kb = ks * 32 + kgA;
            for (int j = 0; j < 8; ++j) {
                af[j]     = pb[rowP * TPAD + kb + j];
                af[j + 8] = pb[rowP * TPAD + kb + 8 + j];
            }
            v16bf bfrag = *(const v16bf*)(Vb + (size_t)colD * TPAD + ks * 32 + kgB);
            acc = __builtin_amdgcn_wmma_f32_16x16x32_bf16(false, af, false, bfrag,
                                                          (short)0, acc, false, false);
        }
        int rb = (lane >> 4) * 8;
        int d  = dt * 16 + (lane & 15);
        if (d < HD) {
            for (int r = 0; r < 8; ++r) {
                int t = t0 + rb + r;
                if (t < T_LEN)
                    Out[((size_t)t * B_SZ + b) * DM + hh * HD + d] = acc[r];
            }
        }
    }
}

// ======================================================================
// residual add + LayerNorm over last dim 720 (one block per token)
// ======================================================================
__global__ __launch_bounds__(256)
void add_ln_kernel(const float* __restrict__ X, const float* __restrict__ A,
                   const float* __restrict__ g, const float* __restrict__ bt,
                   float* __restrict__ Y) {
    __shared__ float red[256];
    int m = blockIdx.x, tid = threadIdx.x;
    const float* x = X + (size_t)m * DM;
    const float* a = A + (size_t)m * DM;
    float s = 0.f, sq = 0.f;
    for (int j = tid; j < DM; j += 256) { float v = x[j] + a[j]; s += v; sq += v * v; }
    float mean = block_reduce_sum256(s, red) / (float)DM;
    float var  = block_reduce_sum256(sq, red) / (float)DM - mean * mean;
    float rstd = rsqrtf(var + 1e-5f);
    for (int j = tid; j < DM; j += 256) {
        float v = x[j] + a[j];
        Y[(size_t)m * DM + j] = (v - mean) * rstd * g[j] + bt[j];
    }
}

// ======================================================================
// Graph propagation: one block (256 threads) per batch element.
// ======================================================================
__global__ __launch_bounds__(256)
void graph_kernel(const float* __restrict__ G0, float* __restrict__ G1, float* __restrict__ G2,
                  const float* __restrict__ atts, float* __restrict__ alpha_all) {
    int b = blockIdx.x, tid = threadIdx.x;
    int wv = tid >> 5, ln = tid & 31;
    __shared__ float s_a[D_INP], s_b[D_INP];
    __shared__ float s_score[E0], s_alpha[E0];
    __shared__ float s_mx[D_INP], s_den[D_INP];
    __shared__ float s_val[2048];
    __shared__ int   s_idx[2048];
    __shared__ int   s_src[E1], s_dst[E1];
    __shared__ float s_w[E1];
    const float* x0 = G0 + (size_t)b * D_INP * CC;
    float* y1 = G1 + (size_t)b * D_INP * CC;
    float* y2 = G2 + (size_t)b * D_INP * CC;

    // ---------- layer 0 (full 36x36 graph; edge e: src=e/36, dst=e%36) ----------
    for (int n = wv; n < D_INP; n += 8) {
        float pa = 0.f, pb2 = 0.f;
        for (int f = ln; f < CC; f += 32) { float xv = x0[n * CC + f]; pa += xv * atts[f]; pb2 += xv * atts[CC + f]; }
        for (int off = 16; off; off >>= 1) { pa += __shfl_xor(pa, off, 32); pb2 += __shfl_xor(pb2, off, 32); }
        if (ln == 0) { s_a[n] = pa; s_b[n] = pb2; }
    }
    __syncthreads();
    for (int e = tid; e < E0; e += 256) {
        int src = e / D_INP, dst = e % D_INP;
        s_score[e] = tanhf(s_a[dst] + s_b[src]);           // edge_weight = 1
    }
    __syncthreads();
    if (tid < D_INP) {
        float mx = -1.0e30f;
        for (int k = 0; k < D_INP; ++k) mx = fmaxf(mx, s_score[36 * k + tid]);
        float den = 0.f;
        for (int k = 0; k < D_INP; ++k) den += expf(s_score[36 * k + tid] - mx);
        s_mx[tid] = mx; s_den[tid] = den;
    }
    __syncthreads();
    for (int e = tid; e < E0; e += 256)
        s_alpha[e] = expf(s_score[e] - s_mx[e % D_INP]) / (s_den[e % D_INP] + 1e-16f);
    __syncthreads();
    for (int n = 0; n < D_INP; ++n)
        for (int f = tid; f < CC; f += 256) {
            float acc = 0.f;
            for (int k = 0; k < D_INP; ++k) acc += s_alpha[36 * k + n] * x0[k * CC + f];
            y1[n * CC + f] = fmaxf(acc, 0.f);
        }
    // ---------- top-k (k = 648) via bitonic sort, desc value / asc index ----------
    for (int i = tid; i < 2048; i += 256) {
        s_val[i] = (i < E0) ? s_alpha[i] : -1.0e38f;
        s_idx[i] = (i < E0) ? i : (1 << 30);
    }
    __syncthreads();
    for (int ksz = 2; ksz <= 2048; ksz <<= 1)
        for (int j = ksz >> 1; j > 0; j >>= 1) {
            for (int i = tid; i < 2048; i += 256) {
                int ixj = i ^ j;
                if (ixj > i) {
                    float v1 = s_val[i], v2 = s_val[ixj];
                    int i1 = s_idx[i], i2 = s_idx[ixj];
                    bool firstGreater = (v1 > v2) || (v1 == v2 && i1 < i2);
                    bool descRegion = ((i & ksz) == 0);
                    bool doSwap = descRegion ? !firstGreater : firstGreater;
                    if (doSwap) { s_val[i] = v2; s_val[ixj] = v1; s_idx[i] = i2; s_idx[ixj] = i1; }
                }
            }
            __syncthreads();
        }
    for (int e = tid; e < E1; e += 256) {
        int oe = s_idx[e];
        s_src[e] = oe / D_INP; s_dst[e] = oe % D_INP; s_w[e] = s_val[e];
    }
    __syncthreads();
    // ---------- layer 1 (648 selected edges) ----------
    const float* att1 = atts + 2 * CC;
    for (int n = wv; n < D_INP; n += 8) {
        float pa = 0.f, pb2 = 0.f;
        for (int f = ln; f < CC; f += 32) { float xv = y1[n * CC + f]; pa += xv * att1[f]; pb2 += xv * att1[CC + f]; }
        for (int off = 16; off; off >>= 1) { pa += __shfl_xor(pa, off, 32); pb2 += __shfl_xor(pb2, off, 32); }
        if (ln == 0) { s_a[n] = pa; s_b[n] = pb2; }
    }
    __syncthreads();
    for (int e = tid; e < E1; e += 256)
        s_score[e] = tanhf(s_a[s_dst[e]] + s_b[s_src[e]]) * s_w[e];
    __syncthreads();
    if (tid < D_INP) {
        float mx = -1.0e30f;
        for (int e = 0; e < E1; ++e) if (s_dst[e] == tid) mx = fmaxf(mx, s_score[e]);
        float den = 0.f;
        for (int e = 0; e < E1; ++e) if (s_dst[e] == tid) den += expf(s_score[e] - mx);
        s_mx[tid] = mx; s_den[tid] = den;
    }
    __syncthreads();
    for (int e = tid; e < E1; e += 256) {
        float al = expf(s_score[e] - s_mx[s_dst[e]]) / (s_den[s_dst[e]] + 1e-16f);
        s_alpha[e] = al;
        alpha_all[(size_t)b * E1 + e] = al;
    }
    __syncthreads();
    for (int n = 0; n < D_INP; ++n)
        for (int f = tid; f < CC; f += 256) {
            float acc = 0.f;
            for (int e = 0; e < E1; ++e) if (s_dst[e] == n) acc += s_alpha[e] * y1[s_src[e] * CC + f];
            y2[n * CC + f] = fmaxf(acc, 0.f);
        }
}

// ======================================================================
// masked aggregation over time + outputs
// ======================================================================
__global__ void agg_kernel(const float* __restrict__ src, const float* __restrict__ Xf,
                           float* __restrict__ flatbuf, float* __restrict__ emb_out) {
    int i = blockIdx.x * 256 + threadIdx.x;
    if (i >= B_SZ * DM) return;
    int c = i % DM, b = i / DM;
    int node = c / 20;
    float s = 0.f, l = 0.f;
    for (int t = 0; t < T_LEN; ++t) {
        size_t tb = (size_t)t * B_SZ + b;
        float m = src[tb * 72 + 36 + node];
        s += Xf[tb * DM + c] * (1.f - m);
        l += m;
    }
    float v = s / (l + 1.f);
    flatbuf[(size_t)b * DFIN + c] = v;
    emb_out[(size_t)b * DM + c] = v;          // embedding output
}

__global__ void emb2flat_kernel(const float* __restrict__ embS, float* __restrict__ flatbuf) {
    int i = blockIdx.x * 256 + threadIdx.x;
    if (i >= B_SZ * D_INP) return;
    int b = i / D_INP, n = i % D_INP;
    flatbuf[(size_t)b * DFIN + DM + n] = embS[i];
}

__global__ void probs_kernel(const float* __restrict__ hcls, const float* __restrict__ W2,
                             const float* __restrict__ b2, float* __restrict__ out) {
    int i = blockIdx.x * 256 + threadIdx.x;
    if (i >= B_SZ * 2) return;
    int b = i / 2, c = i % 2;
    float s = b2[c];
    for (int j = 0; j < DFIN; ++j) s += hcls[(size_t)b * DFIN + j] * W2[(size_t)c * DFIN + j];
    out[i] = s;
}

// ======================================================================
// pairwise distance of alpha rows
// ======================================================================
__global__ __launch_bounds__(256)
void dist_partial_kernel(const float* __restrict__ alpha, float* __restrict__ partial) {
    __shared__ float red[256];
    __shared__ float ai[E1];
    int i = blockIdx.x, tid = threadIdx.x;
    for (int k = tid; k < E1; k += 256) ai[k] = alpha[(size_t)i * E1 + k];
    __syncthreads();
    float s = 0.f;
    for (int j = tid; j < B_SZ; j += 256) {
        float ss = 0.f;
        for (int k = 0; k < E1; ++k) { float d = ai[k] - alpha[(size_t)j * E1 + k]; ss += d * d; }
        s += sqrtf(fmaxf(ss, 0.f));
    }
    float tot = block_reduce_sum256(s, red);
    if (tid == 0) partial[i] = tot;
}

__global__ __launch_bounds__(256)
void dist_final_kernel(const float* __restrict__ partial, float* __restrict__ out) {
    __shared__ float red[256];
    float s = 0.f;
    for (int i = threadIdx.x; i < B_SZ; i += 256) s += partial[i];
    float tot = block_reduce_sum256(s, red);
    if (threadIdx.x == 0) out[0] = tot / ((float)B_SZ * (float)B_SZ);
}

// ======================================================================
// host orchestration
// ======================================================================
static inline unsigned grid1d(size_t n) { return (unsigned)((n + 255) / 256); }

extern "C" void kernel_launch(void* const* d_in, const int* in_sizes, int n_in,
                              void* d_out, int out_size, void* d_ws, size_t ws_size,
                              hipStream_t stream) {
    (void)in_sizes; (void)n_in; (void)out_size; (void)ws_size;
    const float* src        = (const float*)d_in[0];
    const float* stat       = (const float*)d_in[1];
    const float* times      = (const float*)d_in[2];
    const int*   lengths    = (const int*)d_in[3];
    const float* R_u        = (const float*)d_in[4];
    const float* emb_W      = (const float*)d_in[5];
    const float* emb_b      = (const float*)d_in[6];
    const float* graph_atts = (const float*)d_in[7];
    // per-layer params: Wq,bq,Wk,bk,Wv,bv,Wo,bo,W1,b1,W2,b2,ln1_g,ln1_b,ln2_g,ln2_b
    const float* mlp_W1 = (const float*)d_in[40];
    const float* mlp_b1 = (const float*)d_in[41];
    const float* mlp_W2 = (const float*)d_in[42];
    const float* mlp_b2 = (const float*)d_in[43];
    float* out = (float*)d_out;   // [0..1023]=probs, [1024]=distance, [1025..]=embedding

    // ---- workspace carve-up (~1.7 GB; assumes ws is sized generously) ----
    size_t off = 0;
    char* base = (char*)d_ws;
    auto alloc = [&](size_t bytes) -> void* {
        void* p = base + off; off = (off + bytes + 255) & ~(size_t)255; return p;
    };
    float* h_buf   = (float*)alloc((size_t)M_TOK * 144 * 4);
    float* G0      = (float*)alloc((size_t)B_SZ * D_INP * CC * 4);
    float* G1      = (float*)alloc((size_t)B_SZ * D_INP * CC * 4);
    float* G2      = (float*)alloc((size_t)B_SZ * D_INP * CC * 4);
    float* gout    = (float*)alloc((size_t)M_TOK * 144 * 4);
    float* alphaA  = (float*)alloc((size_t)B_SZ * E1 * 4);
    float* peB     = (float*)alloc((size_t)M_TOK * 16 * 4);
    float* embS    = (float*)alloc((size_t)B_SZ * D_INP * 4);
    float* X       = (float*)alloc((size_t)M_TOK * DM * 4);
    float* bufT    = (float*)alloc((size_t)M_TOK * DM * 4);
    float* bufH    = (float*)alloc((size_t)M_TOK * NHID * 4);
    bf16_t* Xp     = (bf16_t*)alloc((size_t)M_TOK * KP_DM * 2);
    bf16_t* Hp     = (bf16_t*)alloc((size_t)M_TOK * NHID * 2);
    bf16_t* Qp     = (bf16_t*)alloc((size_t)B_SZ * NHEAD * TPAD * HDPAD * 2);
    bf16_t* Kpa    = (bf16_t*)alloc((size_t)B_SZ * NHEAD * TPAD * HDPAD * 2);
    bf16_t* VTp    = (bf16_t*)alloc((size_t)B_SZ * NHEAD * HDPAD * TPAD * 2);
    bf16_t* Wp     = (bf16_t*)alloc((size_t)DFPAD * DFPAD * 2);
    float* flatbuf = (float*)alloc((size_t)B_SZ * DFIN * 4);
    bf16_t* flatp  = (bf16_t*)alloc((size_t)B_SZ * DFPAD * 2);
    float* hcls    = (float*)alloc((size_t)B_SZ * DFIN * 4);
    float* partial = (float*)alloc((size_t)B_SZ * 4);

    auto gemm = [&](const bf16_t* A, const bf16_t* Bw, const float* bias, float* C,
                    int M, int N, int Kp, int relu) {
        dim3 g((M + 127) / 128, (N + 15) / 16);
        gemm_bf16_nt<<<g, 256, 0, stream>>>(A, Bw, bias, C, M, N, Kp, relu);
    };
    auto pack = [&](const float* s, bf16_t* d, int Msrc, int Mdst, int Ksrc, int Kdst) {
        size_t tot = (size_t)Mdst * Kdst;
        pack_pad2d<<<grid1d(tot), 256, 0, stream>>>(s, d, Msrc, Ksrc, Kdst, tot);
    };

    // ---- 1. sensor embedding + graph propagation ----
    {
        size_t th = (size_t)M_TOK * 144;
        h_kernel<<<grid1d(th), 256, 0, stream>>>(src, R_u, h_buf, th);
        size_t tg = (size_t)B_SZ * D_INP * CC;
        g0_kernel<<<grid1d(tg), 256, 0, stream>>>(h_buf, G0, tg);
        graph_kernel<<<B_SZ, 256, 0, stream>>>(G0, G1, G2, graph_atts, alphaA);
        gout_kernel<<<grid1d(th), 256, 0, stream>>>(G2, gout, th);
    }
    // ---- 2. positional encoding, static embedding, transformer input ----
    {
        size_t tp = (size_t)M_TOK * 16;
        pe_kernel<<<grid1d(tp), 256, 0, stream>>>(times, peB, tp);
        emb_kernel<<<grid1d(B_SZ * D_INP), 256, 0, stream>>>(stat, emb_W, emb_b, embS);
        size_t tx = (size_t)M_TOK * DM;
        buildx_kernel<<<grid1d(tx), 256, 0, stream>>>(gout, peB, X, tx);
    }
    // ---- 3. transformer layers ----
    size_t qk_tot = (size_t)B_SZ * NHEAD * TPAD * HDPAD;
    for (int l = 0; l < 2; ++l) {
        int p = 8 + l * 16;
        const float* Wq = (const float*)d_in[p + 0];  const float* bq = (const float*)d_in[p + 1];
        const float* Wk = (const float*)d_in[p + 2];  const float* bk = (const float*)d_in[p + 3];
        const float* Wv = (const float*)d_in[p + 4];  const float* bv = (const float*)d_in[p + 5];
        const float* Wo = (const float*)d_in[p + 6];  const float* bo = (const float*)d_in[p + 7];
        const float* W1 = (const float*)d_in[p + 8];  const float* b1 = (const float*)d_in[p + 9];
        const float* W2 = (const float*)d_in[p + 10]; const float* b2 = (const float*)d_in[p + 11];
        const float* g1 = (const float*)d_in[p + 12]; const float* be1 = (const float*)d_in[p + 13];
        const float* g2 = (const float*)d_in[p + 14]; const float* be2 = (const float*)d_in[p + 15];

        pack(X, Xp, M_TOK, M_TOK, DM, KP_DM);
        // Q
        pack(Wq, Wp, DM, DM, DM, KP_DM);
        gemm(Xp, Wp, bq, bufT, M_TOK, DM, KP_DM, 0);
        pack_qk<<<grid1d(qk_tot), 256, 0, stream>>>(bufT, Qp, qk_tot);
        // K
        pack(Wk, Wp, DM, DM, DM, KP_DM);
        gemm(Xp, Wp, bk, bufT, M_TOK, DM, KP_DM, 0);
        pack_qk<<<grid1d(qk_tot), 256, 0, stream>>>(bufT, Kpa, qk_tot);
        // V (transposed pack)
        pack(Wv, Wp, DM, DM, DM, KP_DM);
        gemm(Xp, Wp, bv, bufT, M_TOK, DM, KP_DM, 0);
        pack_vt<<<grid1d(qk_tot), 256, 0, stream>>>(bufT, VTp, qk_tot);
        // fused attention -> bufT (attn output)
        attn_kernel<<<dim3(B_SZ * NHEAD, TPAD / 16), 32, 0, stream>>>(Qp, Kpa, VTp, lengths, bufT);
        // output projection: a = attnout @ Wo^T + bo
        pack(bufT, Xp, M_TOK, M_TOK, DM, KP_DM);
        pack(Wo, Wp, DM, DM, DM, KP_DM);
        gemm(Xp, Wp, bo, bufT, M_TOK, DM, KP_DM, 0);
        // x = LN(x + a)
        add_ln_kernel<<<M_TOK, 256, 0, stream>>>(X, bufT, g1, be1, X);
        // FFN
        pack(X, Xp, M_TOK, M_TOK, DM, KP_DM);
        pack(W1, Wp, NHID, NHID, DM, KP_DM);
        gemm(Xp, Wp, b1, bufH, M_TOK, NHID, KP_DM, 1);
        pack(bufH, Hp, M_TOK, M_TOK, NHID, NHID);
        pack(W2, Wp, DM, DM, NHID, NHID);
        gemm(Hp, Wp, b2, bufT, M_TOK, DM, NHID, 0);
        add_ln_kernel<<<M_TOK, 256, 0, stream>>>(X, bufT, g2, be2, X);
    }
    // ---- 4. masked aggregation, embedding output, classifier ----
    agg_kernel<<<grid1d((size_t)B_SZ * DM), 256, 0, stream>>>(src, X, flatbuf, out + 1025);
    emb2flat_kernel<<<grid1d(B_SZ * D_INP), 256, 0, stream>>>(embS, flatbuf);
    pack(flatbuf, flatp, B_SZ, B_SZ, DFIN, DFPAD);
    pack(mlp_W1, Wp, DFIN, DFPAD, DFIN, DFPAD);
    gemm(flatp, Wp, mlp_b1, hcls, B_SZ, DFIN, DFPAD, 1);
    probs_kernel<<<grid1d(B_SZ * 2), 256, 0, stream>>>(hcls, mlp_W2, mlp_b2, out);
    // ---- 5. pairwise alpha distance ----
    dist_partial_kernel<<<B_SZ, 256, 0, stream>>>(alphaA, partial);
    dist_final_kernel<<<1, 256, 0, stream>>>(partial, out + 1024);
}
